// STConv_input_stacom_45964740001800
// MI455X (gfx1250) — matrix operand order repaired and checked by reading, not compiled, not run on hardware
//
#include <hip/hip_runtime.h>
#include <hip/hip_bf16.h>
#include <stdint.h>

// Problem dims (fixed by reference)
#define BB 2
#define XX 200
#define YY 200
#define ZZ 5
#define CC 128
#define TT 3
#define BN_EPS 1e-3f

typedef __attribute__((ext_vector_type(16))) __bf16          v16bf;
typedef __attribute__((ext_vector_type(8)))  float           v8f;
typedef __attribute__((ext_vector_type(8)))  unsigned short  u16x8;
typedef __attribute__((ext_vector_type(16))) unsigned short  u16x16;

__device__ __forceinline__ unsigned short f2bf(float f) {
  unsigned int u = __float_as_uint(f);
  u += 0x7FFFu + ((u >> 16) & 1u);        // round-to-nearest-even
  return (unsigned short)(u >> 16);
}

// ---------------------------------------------------------------------------
// Weight packing: w[tap(27)][cin(128)][cout(128)] fp32 -> bf16 B-fragments.
// Fragment (tap,kc,nt) = 32 lanes x 16 elems, contiguous 1024B:
//   lane l <16 : N = nt*16+l,      K = kc*32 + {0..7,16..23}[j]
//   lane l>=16 : N = nt*16+(l-16), K = kc*32 + {8..15,24..31}[j]
// Matches the CDNA5 32x16 16-bit B-matrix VGPR layout: each lane loads its
// own contiguous 32 bytes.
// ---------------------------------------------------------------------------
__global__ void pack_weights(const float* __restrict__ w,
                             unsigned short* __restrict__ out) {
  int i = blockIdx.x * blockDim.x + threadIdx.x;
  const int NPACK = 27 * 4 * 8 * 32 * 16;
  if (i >= NPACK) return;
  int j    = i & 15;
  int lane = (i >> 4) & 31;
  int nt   = (i >> 9) & 7;
  int kc   = (i >> 12) & 3;
  int tap  = i >> 14;
  int k = kc * 32 + ((lane >= 16) ? 8 : 0) + (j & 7) + ((j >= 8) ? 16 : 0);
  int n = nt * 16 + (lane & 15);
  out[i] = f2bf(w[(tap * CC + k) * CC + n]);
}

// A-fragment from an LDS row of 128 bf16 (per-lane M row already selected).
// CDNA5 16x32 16-bit A layout: lane<16 holds K {0..7,16..23}, lane>=16 holds
// K {8..15,24..31} within the 32-wide chunk -> two 16B LDS loads per lane.
__device__ __forceinline__ v16bf load_a_frag(const unsigned short* row,
                                             int kc, int lane) {
  int ofs = kc * 32 + ((lane >= 16) ? 8 : 0);
  const u16x8* p = (const u16x8*)(row + ofs);
  u16x8 lo = p[0];   // K base + 0..7   (or 8..15)
  u16x8 hi = p[2];   // K base + 16..23 (or 24..31)
  u16x16 cat = __builtin_shufflevector(lo, hi,
      0,1,2,3,4,5,6,7,8,9,10,11,12,13,14,15);
  return __builtin_bit_cast(v16bf, cat);
}

// ---------------------------------------------------------------------------
// Kernel 1: fused st_conv (3x3x3 over X,Y,T per z-slice) + BN + ReLU + mask
//           + sum over t  ->  bev (bf16) ; also emits h_mask = any_t(mask).
// One WG: 16 x-positions at fixed (b,z,y), all 3 t_out, all 128 c_out.
// 8 waves; wave w owns c_out tile w. Three live accumulators (one per t_out)
// give three independent WMMA dependency chains, and each weight B-fragment
// is loaded once per (tap,kc) and reused across all t_out.
// ---------------------------------------------------------------------------
__global__ void __launch_bounds__(256)
st_conv_kernel(const float* __restrict__ x_feat,
               const unsigned char* __restrict__ x_mask,
               const float* __restrict__ prev_feat,
               const unsigned char* __restrict__ prev_mask,
               const unsigned short* __restrict__ wpack,
               const float* __restrict__ g,  const float* __restrict__ be,
               const float* __restrict__ mu, const float* __restrict__ va,
               unsigned short* __restrict__ bev,
               unsigned char* __restrict__ hmask) {
  __shared__ unsigned short sfeat[162 * CC];   // (xi 0..17, yi 0..2, ti 0..2)
  __shared__ unsigned char  smask[TT * 18];    // per-t occupancy along x

  const int x0 = blockIdx.x * 16;
  const int y  = blockIdx.y;
  const int b  = blockIdx.z / ZZ;
  const int z  = blockIdx.z % ZZ;
  const int tid  = threadIdx.x;
  const int lane = tid & 31;
  const int wave = tid >> 5;

  // ---- stage masked inputs to LDS as bf16 (inputs read exactly once) ----
  for (int r = wave; r < 162; r += 8) {       // row = (xi*3+yi)*3+ti
    int ti = r % 3;
    int yi = (r / 3) % 3;
    int xi = r / 9;
    int x  = x0 + xi - 1;
    int y2 = y + yi - 1;
    float4 v = make_float4(0.f, 0.f, 0.f, 0.f);
    float m = 0.f;
    unsigned char mb = 0;
    if (x >= 0 && x < XX && y2 >= 0 && y2 < YY) {
      size_t vox = ((size_t)((b * XX + x) * YY + y2)) * ZZ + z;
      const float* src;
      if (ti == 0) { src = x_feat + vox * CC; mb = x_mask[vox]; }
      else {
        size_t pvox = (size_t)(ti - 1) * (BB * XX * YY * ZZ) + vox;
        src = prev_feat + pvox * CC; mb = prev_mask[pvox];
      }
      m = mb ? 1.f : 0.f;
      v = ((const float4*)src)[lane];
    }
    unsigned short* d = sfeat + r * CC + lane * 4;
    d[0] = f2bf(v.x * m); d[1] = f2bf(v.y * m);
    d[2] = f2bf(v.z * m); d[3] = f2bf(v.w * m);
    if (yi == 1 && lane == 0) smask[ti * 18 + xi] = mb;
  }
  __syncthreads();

  // ---- implicit GEMM with WMMA ----
  const int nt = wave;
  const int ch = nt * 16 + (lane & 15);        // D layout: lane -> N column
  const float scale = g[ch] * rsqrtf(va[ch] + BN_EPS);
  const float shift = be[ch] - mu[ch] * scale;
  const int mrow = lane & 15;                  // A layout: lane -> M row

  v8f acc[TT];
#pragma unroll
  for (int t = 0; t < TT; ++t) acc[t] = (v8f){};

  for (int dx = 0; dx < 3; ++dx)
    for (int dy = 0; dy < 3; ++dy) {
      // LDS base for this (dx,dy): rows for ti follow at stride CC
      const unsigned short* abase =
          sfeat + (size_t)(((mrow + dx) * 3 + dy) * 3) * CC;
      for (int dt = 0; dt < 3; ++dt) {
        int tap = (dx * 3 + dy) * 3 + dt;      // st_w is [dx][dy][dt][ci][co]
#pragma unroll
        for (int kc = 0; kc < 4; ++kc) {
          const u16x16* wp = (const u16x16*)
              (wpack + (size_t)((tap * 4 + kc) * 8 + nt) * 512);
          v16bf bmat = __builtin_bit_cast(v16bf, wp[lane]);  // load once
#pragma unroll
          for (int tout = 0; tout < TT; ++tout) {            // reuse 2-3x
            int ti = tout + dt - 1;
            if (ti < 0 || ti >= TT) continue;  // zero-padded T boundary
            v16bf a = load_a_frag(abase + (size_t)ti * CC, kc, lane);
            acc[tout] = __builtin_amdgcn_wmma_f32_16x16x32_bf16(
                false, a, false, bmat, (short)0, acc[tout], false, false);
          }
        }
      }
    }

  // ---- BN -> ReLU -> per-(x,t) mask -> t-sum -> store bev (bf16) ----
#pragma unroll
  for (int r = 0; r < 8; ++r) {
    int m = r + ((lane >= 16) ? 8 : 0);        // D layout: VGPR r -> M row
    float s = 0.f;
#pragma unroll
    for (int tout = 0; tout < TT; ++tout) {
      float mk = smask[tout * 18 + m + 1] ? 1.f : 0.f;
      float vv = acc[tout][r] * scale + shift;
      vv = vv > 0.f ? vv : 0.f;
      s += vv * mk;
    }
    int x = x0 + m;
    if (x < XX) {
      size_t idx = (((size_t)((b * XX + x) * YY + y)) * ZZ + z) * CC + ch;
      bev[idx] = f2bf(s);
    }
  }
  if (tid < 16) {
    int x = x0 + tid;
    if (x < XX) {
      unsigned char hm = smask[tid + 1] | smask[18 + tid + 1] | smask[36 + tid + 1];
      hmask[((size_t)((b * XX + x) * YY + y)) * ZZ + z] = hm ? 1 : 0;
    }
  }
}

// ---------------------------------------------------------------------------
// Kernel 2: height_conv (3x3x3 over X,Y,Z) + BN + ReLU + h_mask -> out fp32.
// One WG: 16 x-positions at fixed (b,y), all 5 z_out, all 128 c_out.
// Five live accumulators (one per z_out); each B-fragment loaded once per
// (tap,kc) and reused across all valid z_out.
// ---------------------------------------------------------------------------
__global__ void __launch_bounds__(256)
h_conv_kernel(const unsigned short* __restrict__ bev,
              const unsigned char* __restrict__ hmask,
              const unsigned short* __restrict__ wpack,
              const float* __restrict__ g,  const float* __restrict__ be,
              const float* __restrict__ mu, const float* __restrict__ va,
              float* __restrict__ out) {
  __shared__ unsigned short sfeat[270 * CC];   // (xi 0..17, yi 0..2, zi 0..4)
  __shared__ unsigned char  smask[ZZ * 18];

  const int x0 = blockIdx.x * 16;
  const int y  = blockIdx.y;
  const int b  = blockIdx.z;
  const int tid  = threadIdx.x;
  const int lane = tid & 31;
  const int wave = tid >> 5;

  for (int r = wave; r < 270; r += 8) {        // row = (xi*3+yi)*5+zi
    int zi = r % 5;
    int yi = (r / 5) % 3;
    int xi = r / 15;
    int x  = x0 + xi - 1;
    int y2 = y + yi - 1;
    uint2 v = make_uint2(0u, 0u);              // 4 bf16
    unsigned char mb = 0;
    if (x >= 0 && x < XX && y2 >= 0 && y2 < YY) {
      size_t vox = ((size_t)((b * XX + x) * YY + y2)) * ZZ + zi;
      v  = ((const uint2*)(bev + vox * CC))[lane];
      mb = hmask[vox];
    }
    ((uint2*)(sfeat + (size_t)r * CC))[lane] = v;
    if (yi == 1 && lane == 0) smask[zi * 18 + xi] = mb;
  }
  __syncthreads();

  const int nt = wave;
  const int ch = nt * 16 + (lane & 15);
  const float scale = g[ch] * rsqrtf(va[ch] + BN_EPS);
  const float shift = be[ch] - mu[ch] * scale;
  const int mrow = lane & 15;

  v8f acc[ZZ];
#pragma unroll
  for (int zo = 0; zo < ZZ; ++zo) acc[zo] = (v8f){};

  for (int dx = 0; dx < 3; ++dx)
    for (int dy = 0; dy < 3; ++dy) {
      const unsigned short* abase =
          sfeat + (size_t)(((mrow + dx) * 3 + dy) * 5) * CC;
      for (int dz = 0; dz < 3; ++dz) {
        int tap = (dx * 3 + dy) * 3 + dz;      // h_w is [dx][dy][dz][ci][co]
#pragma unroll
        for (int kc = 0; kc < 4; ++kc) {
          const u16x16* wp = (const u16x16*)
              (wpack + (size_t)((tap * 4 + kc) * 8 + nt) * 512);
          v16bf bmat = __builtin_bit_cast(v16bf, wp[lane]);  // load once
#pragma unroll
          for (int zo = 0; zo < ZZ; ++zo) {                  // reuse up to 5x
            int zi = zo + dz - 1;
            if (zi < 0 || zi >= ZZ) continue;  // zero-padded Z boundary
            v16bf a = load_a_frag(abase + (size_t)zi * CC, kc, lane);
            acc[zo] = __builtin_amdgcn_wmma_f32_16x16x32_bf16(
                false, a, false, bmat, (short)0, acc[zo], false, false);
          }
        }
      }
    }

#pragma unroll
  for (int zo = 0; zo < ZZ; ++zo) {
#pragma unroll
    for (int r = 0; r < 8; ++r) {
      int m = r + ((lane >= 16) ? 8 : 0);
      int x = x0 + m;
      if (x < XX) {
        float mk = smask[zo * 18 + m + 1] ? 1.f : 0.f;
        float vv = acc[zo][r] * scale + shift;
        vv = vv > 0.f ? vv : 0.f;
        out[(((size_t)((b * XX + x) * YY + y)) * ZZ + zo) * CC + ch] = vv * mk;
      }
    }
  }
}

// ---------------------------------------------------------------------------
extern "C" void kernel_launch(void* const* d_in, const int* in_sizes, int n_in,
                              void* d_out, int out_size, void* d_ws, size_t ws_size,
                              hipStream_t stream) {
  (void)in_sizes; (void)n_in; (void)out_size; (void)ws_size;

  const float*         x_feat    = (const float*)d_in[0];
  const unsigned char* x_mask    = (const unsigned char*)d_in[1];
  const float*         prev_feat = (const float*)d_in[2];
  const unsigned char* prev_mask = (const unsigned char*)d_in[3];
  const float* st_w = (const float*)d_in[4];
  const float* st_g = (const float*)d_in[5];
  const float* st_b = (const float*)d_in[6];
  const float* st_m = (const float*)d_in[7];
  const float* st_v = (const float*)d_in[8];
  const float* h_w  = (const float*)d_in[9];
  const float* h_g  = (const float*)d_in[10];
  const float* h_b  = (const float*)d_in[11];
  const float* h_m  = (const float*)d_in[12];
  const float* h_v  = (const float*)d_in[13];

  // Workspace layout (all offsets 32B-aligned):
  //   [0)        wpack_st : 27*128*128 bf16 = 884736 B
  //   [884736)   wpack_h  : 884736 B
  //   [1769472)  h_mask   : 400000 B (padded to 401408)
  //   [2170880)  bev bf16 : 2*200*200*5*128*2 = 102.4 MB
  char* ws = (char*)d_ws;
  const size_t WPACK_ELEMS = 27u * 128u * 128u;
  unsigned short* wp_st = (unsigned short*)(ws);
  unsigned short* wp_h  = (unsigned short*)(ws + WPACK_ELEMS * 2);
  unsigned char*  hmask = (unsigned char*)(ws + 2 * WPACK_ELEMS * 2);
  unsigned short* bev   = (unsigned short*)(ws + 2 * WPACK_ELEMS * 2 + 401408);

  const int packBlocks = (int)((WPACK_ELEMS + 255) / 256);
  pack_weights<<<packBlocks, 256, 0, stream>>>(st_w, wp_st);
  pack_weights<<<packBlocks, 256, 0, stream>>>(h_w,  wp_h);

  dim3 g1((XX + 15) / 16, YY, BB * ZZ);
  st_conv_kernel<<<g1, 256, 0, stream>>>(x_feat, x_mask, prev_feat, prev_mask,
                                         wp_st, st_g, st_b, st_m, st_v,
                                         bev, hmask);

  dim3 g2((XX + 15) / 16, YY, BB);
  h_conv_kernel<<<g2, 256, 0, stream>>>(bev, hmask, wp_h,
                                        h_g, h_b, h_m, h_v, (float*)d_out);
}
